// SimpleMLP_44341242364599
// MI455X (gfx1250) — matrix operand order repaired and checked by reading, not compiled, not run on hardware
//
#include <hip/hip_runtime.h>
#include <hip/hip_bf16.h>
#include <stdint.h>

typedef __bf16 bf16_t;
typedef __bf16 v8bf  __attribute__((ext_vector_type(8)));
typedef __bf16 v16bf __attribute__((ext_vector_type(16)));
typedef float  v8f   __attribute__((ext_vector_type(8)));
typedef float  v4f   __attribute__((ext_vector_type(4)));
typedef unsigned int u32x4 __attribute__((ext_vector_type(4)));
typedef int          i32x4 __attribute__((ext_vector_type(4)));
typedef int          i32x8 __attribute__((ext_vector_type(8)));

#define D_MODEL 4096
#define BATCH   8192
#define N_HEADS 32
#define D_K     128

#if defined(__HIP_DEVICE_COMPILE__) && __has_builtin(__builtin_amdgcn_tensor_load_to_lds)
#define USE_TDM 1
#else
#define USE_TDM 0
#endif

// ---------------------------------------------------------------------------
// f32 -> bf16 conversion (bandwidth-bound, runs once per buffer)
// ---------------------------------------------------------------------------
__global__ __launch_bounds__(256)
void f32_to_bf16_kernel(const float* __restrict__ src, bf16_t* __restrict__ dst,
                        long n) {
    long i = ((long)blockIdx.x * 256 + threadIdx.x) * 8;
    if (i + 8 <= n) {
        v4f a = *(const v4f*)(src + i);
        v4f b = *(const v4f*)(src + i + 4);
        v8bf o;
        o[0] = (bf16_t)a[0]; o[1] = (bf16_t)a[1];
        o[2] = (bf16_t)a[2]; o[3] = (bf16_t)a[3];
        o[4] = (bf16_t)b[0]; o[5] = (bf16_t)b[1];
        o[6] = (bf16_t)b[2]; o[7] = (bf16_t)b[3];
        *(v8bf*)(dst + i) = o;
    }
}

// ---------------------------------------------------------------------------
// TDM descriptor: async-copy one [128 rows x 32 K] bf16 tile (row-major,
// row stride = 4096 elems) from global memory into LDS. The TDM pad feature
// inserts 16B after every 64B row -> LDS row stride 80B (= LDSA_STRIDE elems),
// matching the WMMA fragment addressing below.
// ---------------------------------------------------------------------------
#define LDSA_STRIDE 40   // 32 + 8 pad elems: 80B rows, 16B aligned, b128 conflict-free

#if USE_TDM
__device__ __forceinline__ void tdm_load_tile(const bf16_t* gbase,
                                              size_t elem_off,
                                              unsigned lds_byte,
                                              int rows_total) {
    const unsigned long long ga =
        (unsigned long long)(uintptr_t)gbase + (unsigned long long)elem_off * 2ull;

    u32x4 g0;
    g0[0] = 1u;                                        // count=1 (valid), user mode
    g0[1] = lds_byte;                                  // lds_addr
    g0[2] = (unsigned)(ga & 0xffffffffull);            // global_addr[31:0]
    g0[3] = (unsigned)((ga >> 32) & 0x01ffffffull)     // global_addr[56:32]
          | (2u << 30);                                // D# type = 2 ("image")

    i32x8 g1;
    g1[0] = (1 << 16)                                  // data_size = 2 bytes
          | (1 << 20)                                  // pad_enable
          | (3 << 22)                                  // pad_interval: 16 DWORDs (64B)
          | (3 << 25);                                 // pad_amount:   4 DWORDs (16B)
    g1[1] = (int)((4096u & 0xffffu) << 16);            // tensor_dim0[15:0] (=4096)
    g1[2] = (int)(((unsigned)rows_total & 0xffffu) << 16);  // dim0[31:16]=0 | dim1[15:0]
    g1[3] = (int)((32u << 16) | (((unsigned)rows_total >> 16) & 0xffffu)); // dim1 hi | tile_dim0=32
    g1[4] = 128;                                       // tile_dim1 = 128 rows (tile_dim2 = 0)
    g1[5] = 4096;                                      // tensor_dim0_stride[31:0]
    g1[6] = 0;                                         // stride hi / dim1_stride lo
    g1[7] = 0;

    const i32x4 gz4 = {0, 0, 0, 0};
    const i32x8 gz8 = {0, 0, 0, 0, 0, 0, 0, 0};
    // 6-arg form (clang-23 / therock-10.0 headers): extra int32x8 group,
    // zero-filled here (2D tensor: groups 2/3 unused).
    __builtin_amdgcn_tensor_load_to_lds(g0, g1, gz4, gz4, gz8, 0);
}
#endif

// ---------------------------------------------------------------------------
// bf16 WMMA GEMM:  C[M,N] = A[M,K] * B[N,K]^T   (torch Linear convention)
// Block: 256 threads = 8 waves. Block tile 128x128, K-step 32, double-buffered
// LDS filled by the Tensor Data Mover (prefetch next tile during compute).
// Wave grid 4x2 (M x N): each wave computes 32x64 = 2x4 WMMA tiles of 16x16.
// FUSE=false : store bf16 (Q/K/V path)
// FUSE=true  : store f32 = C + bias[col] + residual[row,col]
// ---------------------------------------------------------------------------
template <bool FUSE>
__global__ __launch_bounds__(256)
void gemm_bf16_wmma(const bf16_t* __restrict__ A,
                    const bf16_t* __restrict__ B,
                    bf16_t* __restrict__ Cbf,
                    float*  __restrict__ Cf,
                    const float* __restrict__ bias,
                    const float* __restrict__ residual,
                    int M, int N, int K) {
    __shared__ bf16_t lA[2][128 * LDSA_STRIDE];
    __shared__ bf16_t lB[2][128 * LDSA_STRIDE];

    const int tid  = threadIdx.x;
    const int lane = tid & 31;
    const int wid  = tid >> 5;
    const int wm   = (wid & 3) * 32;   // wave M offset inside block tile
    const int wn   = (wid >> 2) * 64;  // wave N offset inside block tile
    const int bm   = blockIdx.y * 128;
    const int bn   = blockIdx.x * 128;

    v8f acc[2][4] = {};

#if USE_TDM
    const unsigned ldsA[2] = {(unsigned)(uintptr_t)&lA[0][0],
                              (unsigned)(uintptr_t)&lA[1][0]};
    const unsigned ldsB[2] = {(unsigned)(uintptr_t)&lB[0][0],
                              (unsigned)(uintptr_t)&lB[1][0]};
    if (wid == 0) {                     // one wave drives the TDM engine
        tdm_load_tile(A, (size_t)bm * K, ldsA[0], M);
        tdm_load_tile(B, (size_t)bn * K, ldsB[0], N);
    }
#else
    const int lr = tid >> 2;            // 0..63
    const int lk = (tid & 3) * 8;       // 0,8,16,24
#endif

    for (int k0 = 0; k0 < K; k0 += 32) {
        const int cur = (k0 >> 5) & 1;

#if USE_TDM
        if (wid == 0) {
            if (k0 + 32 < K) {          // prefetch next tile into other buffer
                tdm_load_tile(A, (size_t)bm * K + (k0 + 32), ldsA[cur ^ 1], M);
                tdm_load_tile(B, (size_t)bn * K + (k0 + 32), ldsB[cur ^ 1], N);
                // TDM ops complete in order per wave: <=2 pending means the
                // current tile's two descriptors have fully landed in LDS.
                __builtin_amdgcn_s_wait_tensorcnt((short)2);
            } else {
                __builtin_amdgcn_s_wait_tensorcnt((short)0);
            }
        }
        __syncthreads();                // publish TDM-written LDS to all waves
#else
        *(v8bf*)(&lA[cur][(lr     ) * LDSA_STRIDE + lk]) =
            *(const v8bf*)(A + (size_t)(bm + lr     ) * K + k0 + lk);
        *(v8bf*)(&lA[cur][(lr + 64) * LDSA_STRIDE + lk]) =
            *(const v8bf*)(A + (size_t)(bm + lr + 64) * K + k0 + lk);
        *(v8bf*)(&lB[cur][(lr     ) * LDSA_STRIDE + lk]) =
            *(const v8bf*)(B + (size_t)(bn + lr     ) * K + k0 + lk);
        *(v8bf*)(&lB[cur][(lr + 64) * LDSA_STRIDE + lk]) =
            *(const v8bf*)(B + (size_t)(bn + lr + 64) * K + k0 + lk);
        __syncthreads();
#endif

        const bf16_t* __restrict__ bufA = &lA[cur][0];
        const bf16_t* __restrict__ bufB = &lB[cur][0];

        // A fragment: lane<16 -> M=lane, K elems [0..7] & [16..23];
        //             lane>=16 -> M=lane-16, K elems [8..15] & [24..31]
        const int mrow = wm + (lane & 15);
        const int akb  = (lane >> 4) * 8;
        v16bf afrag[2];
#pragma unroll
        for (int tm = 0; tm < 2; ++tm) {
            const bf16_t* p = bufA + (mrow + tm * 16) * LDSA_STRIDE + akb;
            v8bf lo = *(const v8bf*)(p);
            v8bf hi = *(const v8bf*)(p + 16);
            afrag[tm] = __builtin_shufflevector(lo, hi,
                0,1,2,3,4,5,6,7,8,9,10,11,12,13,14,15);
        }
        // B fragment: lane<16 -> N=lane, K=[0..7] & [16..23]; lane>=16 -> K=[8..15] & [24..31]
        const int ncol = wn + (lane & 15);
        const int bkb  = (lane >> 4) * 8;
        v16bf bfrag[4];
#pragma unroll
        for (int tn = 0; tn < 4; ++tn) {
            const bf16_t* p = bufB + (ncol + tn * 16) * LDSA_STRIDE + bkb;
            v8bf lo = *(const v8bf*)(p);
            v8bf hi = *(const v8bf*)(p + 16);
            bfrag[tn] = __builtin_shufflevector(lo, hi,
                0,1,2,3,4,5,6,7,8,9,10,11,12,13,14,15);
        }
#pragma unroll
        for (int tm = 0; tm < 2; ++tm)
#pragma unroll
            for (int tn = 0; tn < 4; ++tn)
                acc[tm][tn] = __builtin_amdgcn_wmma_f32_16x16x32_bf16(
                    false, afrag[tm], false, bfrag[tn],
                    (short)0, acc[tm][tn], false, false);
        __syncthreads();                // buffer may be overwritten next iter
    }

    // C layout: vgpr r, lanes 0-15: (M=r, N=lane); lanes 16-31: (M=r+8, N=lane-16)
    const int rbase = bm + wm + (lane >> 4) * 8;
    const int cbase = bn + wn + (lane & 15);
#pragma unroll
    for (int tm = 0; tm < 2; ++tm) {
#pragma unroll
        for (int tn = 0; tn < 4; ++tn) {
            const int col = cbase + tn * 16;
#pragma unroll
            for (int r = 0; r < 8; ++r) {
                const int row = rbase + tm * 16 + r;
                const float v = acc[tm][tn][r];
                if constexpr (FUSE) {
                    Cf[(size_t)row * N + col] =
                        v + bias[col] + residual[(size_t)row * N + col];
                } else {
                    Cbf[(size_t)row * N + col] = (bf16_t)v;
                }
            }
        }
    }
}

// ---------------------------------------------------------------------------
// Per-row head attention: one 256-thread block per batch row.
// Q,K,V: [BATCH, 32 heads, 128] bf16 -> out bf16. All math in LDS/f32.
// ---------------------------------------------------------------------------
__global__ __launch_bounds__(256)
void attention_kernel(const bf16_t* __restrict__ Q,
                      const bf16_t* __restrict__ Kh,
                      const bf16_t* __restrict__ V,
                      bf16_t* __restrict__ out) {
    __shared__ float sq[D_MODEL];
    __shared__ float sk[D_MODEL];
    __shared__ float sv[D_MODEL];
    __shared__ float sp[N_HEADS * N_HEADS];

    const size_t base = (size_t)blockIdx.x * D_MODEL;
    for (int i = threadIdx.x; i < D_MODEL; i += 256) {
        sq[i] = (float)Q[base + i];
        sk[i] = (float)Kh[base + i];
        sv[i] = (float)V[base + i];
    }
    __syncthreads();

    const float scale = 0.08838834764831845f;  // 1/sqrt(128)
    for (int s = threadIdx.x; s < N_HEADS * N_HEADS; s += 256) {
        const int h = s >> 5, g = s & 31;
        const float* q  = &sq[h * D_K];
        const float* kk = &sk[g * D_K];
        float a = 0.0f;
#pragma unroll 8
        for (int d = 0; d < D_K; ++d) a = fmaf(q[d], kk[d], a);
        sp[s] = a * scale;
    }
    __syncthreads();

    if (threadIdx.x < N_HEADS) {
        float* row = &sp[threadIdx.x * N_HEADS];
        float m = row[0];
        for (int g = 1; g < N_HEADS; ++g) m = fmaxf(m, row[g]);
        float sum = 0.0f;
        for (int g = 0; g < N_HEADS; ++g) {
            const float e = __expf(row[g] - m);
            row[g] = e;
            sum += e;
        }
        const float inv = 1.0f / sum;
        for (int g = 0; g < N_HEADS; ++g) row[g] *= inv;
    }
    __syncthreads();

    for (int o = threadIdx.x; o < D_MODEL; o += 256) {
        const int h = o >> 7, d = o & 127;
        const float* p = &sp[h * N_HEADS];
        float a = 0.0f;
#pragma unroll
        for (int g = 0; g < N_HEADS; ++g) a = fmaf(p[g], sv[g * D_K + d], a);
        out[base + o] = (bf16_t)a;
    }
}

// ---------------------------------------------------------------------------
// In-place LayerNorm over rows of 4096 f32 (one 256-thread block per row).
// ---------------------------------------------------------------------------
__global__ __launch_bounds__(256)
void layernorm_kernel(float* __restrict__ y,
                      const float* __restrict__ gamma,
                      const float* __restrict__ beta) {
    __shared__ float s1[256];
    __shared__ float s2[256];
    const size_t base = (size_t)blockIdx.x * D_MODEL;
    const int tid = threadIdx.x;

    v4f v[4];
    float sum = 0.0f, sq = 0.0f;
#pragma unroll
    for (int c = 0; c < 4; ++c) {
        v[c] = *(const v4f*)(y + base + tid * 16 + c * 4);
#pragma unroll
        for (int j = 0; j < 4; ++j) {
            sum += v[c][j];
            sq  = fmaf(v[c][j], v[c][j], sq);
        }
    }
    s1[tid] = sum;
    s2[tid] = sq;
    __syncthreads();
    for (int off = 128; off > 0; off >>= 1) {
        if (tid < off) { s1[tid] += s1[tid + off]; s2[tid] += s2[tid + off]; }
        __syncthreads();
    }
    const float mu  = s1[0] * (1.0f / D_MODEL);
    const float var = s2[0] * (1.0f / D_MODEL) - mu * mu;
    const float inv = rsqrtf(var + 1e-5f);

#pragma unroll
    for (int c = 0; c < 4; ++c) {
        v4f o;
#pragma unroll
        for (int j = 0; j < 4; ++j) {
            const int col = tid * 16 + c * 4 + j;
            o[j] = (v[c][j] - mu) * inv * gamma[col] + beta[col];
        }
        *(v4f*)(y + base + tid * 16 + c * 4) = o;
    }
}

// ---------------------------------------------------------------------------
// Host-side orchestration
// ---------------------------------------------------------------------------
extern "C" void kernel_launch(void* const* d_in, const int* in_sizes, int n_in,
                              void* d_out, int out_size, void* d_ws, size_t ws_size,
                              hipStream_t stream) {
    const float* x     = (const float*)d_in[0];
    const float* wq    = (const float*)d_in[1];
    const float* wk    = (const float*)d_in[2];
    const float* wv    = (const float*)d_in[3];
    const float* wo    = (const float*)d_in[4];
    const float* bo    = (const float*)d_in[5];
    const float* gamma = (const float*)d_in[6];
    const float* beta  = (const float*)d_in[7];
    float* out = (float*)d_out;

    const long nX = (long)BATCH * D_MODEL;          // 33,554,432
    const long nW = (long)D_MODEL * D_MODEL;        // 16,777,216

    char* ws = (char*)d_ws;
    size_t off = 0;
    bf16_t* xb    = (bf16_t*)(ws + off); off += (size_t)nX * 2;
    bf16_t* wqb   = (bf16_t*)(ws + off); off += (size_t)nW * 2;
    bf16_t* wkb   = (bf16_t*)(ws + off); off += (size_t)nW * 2;
    bf16_t* wvb   = (bf16_t*)(ws + off); off += (size_t)nW * 2;
    bf16_t* wob   = (bf16_t*)(ws + off); off += (size_t)nW * 2;
    bf16_t* Qb    = (bf16_t*)(ws + off); off += (size_t)nX * 2;
    bf16_t* Kb    = (bf16_t*)(ws + off); off += (size_t)nX * 2;
    bf16_t* Vb    = (bf16_t*)(ws + off); off += (size_t)nX * 2;
    bf16_t* attnb = (bf16_t*)(ws + off); off += (size_t)nX * 2;

    // 1) precision downconvert (one-shot, bandwidth-bound)
    f32_to_bf16_kernel<<<dim3(nX / 2048), dim3(256), 0, stream>>>(x,  xb,  nX);
    f32_to_bf16_kernel<<<dim3(nW / 2048), dim3(256), 0, stream>>>(wq, wqb, nW);
    f32_to_bf16_kernel<<<dim3(nW / 2048), dim3(256), 0, stream>>>(wk, wkb, nW);
    f32_to_bf16_kernel<<<dim3(nW / 2048), dim3(256), 0, stream>>>(wv, wvb, nW);
    f32_to_bf16_kernel<<<dim3(nW / 2048), dim3(256), 0, stream>>>(wo, wob, nW);

    // 2) Q/K/V projections (compute-bound bf16 WMMA GEMMs, TDM-fed LDS)
    dim3 ggrid(D_MODEL / 128, BATCH / 128);  // (32, 64)
    gemm_bf16_wmma<false><<<ggrid, 256, 0, stream>>>(
        xb, wqb, Qb, nullptr, nullptr, nullptr, BATCH, D_MODEL, D_MODEL);
    gemm_bf16_wmma<false><<<ggrid, 256, 0, stream>>>(
        xb, wkb, Kb, nullptr, nullptr, nullptr, BATCH, D_MODEL, D_MODEL);
    gemm_bf16_wmma<false><<<ggrid, 256, 0, stream>>>(
        xb, wvb, Vb, nullptr, nullptr, nullptr, BATCH, D_MODEL, D_MODEL);

    // 3) per-row head attention
    attention_kernel<<<dim3(BATCH), dim3(256), 0, stream>>>(Qb, Kb, Vb, attnb);

    // 4) output projection fused with bias + residual (f32 out)
    gemm_bf16_wmma<true><<<ggrid, 256, 0, stream>>>(
        attnb, wob, nullptr, out, bo, x, BATCH, D_MODEL, D_MODEL);

    // 5) in-place LayerNorm
    layernorm_kernel<<<dim3(BATCH), dim3(256), 0, stream>>>(out, gamma, beta);
}
